// MyGCN_15564961480907
// MI455X (gfx1250) — matrix operand order repaired and checked by reading, not compiled, not run on hardware
//
#include <hip/hip_runtime.h>
#include <hip/hip_bf16.h>

// ---------------------------------------------------------------------------
// Types for CDNA5 WMMA (wave32, 16x16x32 bf16 -> f32)
// ---------------------------------------------------------------------------
typedef __attribute__((ext_vector_type(16))) __bf16 v16bf;
typedef __attribute__((ext_vector_type(8)))  __bf16 v8bf;
typedef __attribute__((ext_vector_type(8)))  float  v8f;

#define NEG_SLOPE 0.2f

__device__ __forceinline__ unsigned short f2bf(float f) {
  unsigned int u = __float_as_uint(f);
  u += 0x7FFFu + ((u >> 16) & 1u);           // round-to-nearest-even
  return (unsigned short)(u >> 16);
}

// ---------------------------------------------------------------------------
// Elementwise / utility kernels
// ---------------------------------------------------------------------------
// converts f32 [rows,K] -> bf16 [rowsPad,Kp], zero padded in both dims
__global__ void k_f32_to_bf16_pad(const float* __restrict__ s,
                                  unsigned short* __restrict__ d,
                                  long long rows, long long rowsPad,
                                  int K, int Kp) {
  long long total = rowsPad * (long long)Kp;
  for (long long i = blockIdx.x * (long long)blockDim.x + threadIdx.x;
       i < total; i += (long long)gridDim.x * blockDim.x) {
    long long r = i / Kp;
    int c = (int)(i - r * Kp);
    d[i] = (r < rows && c < K) ? f2bf(s[r * K + c]) : (unsigned short)0;
  }
}

__global__ void k_zero_f32(float* __restrict__ p, long long n) {
  for (long long i = blockIdx.x * (long long)blockDim.x + threadIdx.x;
       i < n; i += (long long)gridDim.x * blockDim.x)
    p[i] = 0.0f;
}

__global__ void k_fill_f32(float* __restrict__ p, float v, long long n) {
  for (long long i = blockIdx.x * (long long)blockDim.x + threadIdx.x;
       i < n; i += (long long)gridDim.x * blockDim.x)
    p[i] = v;
}

// ---------------------------------------------------------------------------
// WMMA GEMM: C[M,Npad] = A[M,Kp] * B[Npad,Kp]^T  (A,B bf16 row-major, C f32)
// Requires: Npad % 128 == 0, Kp % 64 == 0. M arbitrary (clamped/guarded).
// Block: 256 threads = 8 waves (2x4); workgroup tile 64(M) x 128(N);
// each wave holds a 32x32 register tile = 2x2 WMMA accumulators.
// Double-buffered LDS staging via CDNA5 async copies (ASYNCcnt pipeline):
// stage s+1 loads overlap stage s compute; one barrier per stage.
// ---------------------------------------------------------------------------
#define TM 64
#define TN 128
#define TK 64    // K per LDS stage = 2 WMMA K-steps

__global__ __launch_bounds__(256) void k_gemm_bf16_wmma(
    const unsigned short* __restrict__ A,
    const unsigned short* __restrict__ B,
    float* __restrict__ C,
    int M, int Npad, int Kp) {
  __shared__ unsigned short lA[2][TM * TK];   // 2 x 8KB
  __shared__ unsigned short lB[2][TN * TK];   // 2 x 16KB

  const int mBase = blockIdx.y * TM;
  const int nBase = blockIdx.x * TN;
  const int tid   = threadIdx.x;
  const int lane  = tid & 31;
  const int wave  = tid >> 5;
  const int wm    = wave & 1;    // 2 M-groups of 32
  const int wn    = wave >> 1;   // 4 N-groups of 32
  const int hlf   = lane >> 4;   // lane half (K split)
  const int mrow  = lane & 15;

  // async cooperative-load mapping:
  // A: 4096 bf16/stage -> 16/thread (2 x b128); B: 8192 bf16 -> 32/thread (4 x b128)
  const int aIdx = tid * 16;
  const int aRow = aIdx >> 6;
  const int aCol = aIdx & 63;
  const int gARow = (mBase + aRow < M) ? (mBase + aRow) : (M - 1); // clamp tail
  const int bIdx = tid * 32;
  const int bRow = bIdx >> 6;
  const int bCol = bIdx & 63;

  v8f acc[2][2] = {};

  const int aOff0 = (wm * 32 + mrow) * TK;        // M subtile 0
  const int aOff1 = (wm * 32 + 16 + mrow) * TK;   // M subtile 1
  const int bOff0 = (wn * 32 + mrow) * TK;        // N subtile 0
  const int bOff1 = (wn * 32 + 16 + mrow) * TK;   // N subtile 1

  const int nStages = Kp / TK;

  // Issue CDNA5 async global->LDS copies for one stage into buffer `buf`.
  // Instruction offset applies to both LDS dest and global src (ISA 08 §4.4).
  auto issue_async = [&](int stage, int buf) {
    const unsigned short* gA = &A[(size_t)gARow * Kp + stage * TK + aCol];
    const unsigned short* gB = &B[(size_t)(nBase + bRow) * Kp + stage * TK + bCol];
    unsigned la0 = (unsigned)(size_t)&lA[buf][aRow * TK + aCol];
    unsigned lb0 = (unsigned)(size_t)&lB[buf][bRow * TK + bCol];
    unsigned long long ga = (unsigned long long)(size_t)gA;
    unsigned long long gb = (unsigned long long)(size_t)gB;
    __builtin_prefetch(gA + TK, 0, 1);   // L2 hint for the stage after next
    __builtin_prefetch(gB + TK, 0, 1);
    asm volatile("global_load_async_to_lds_b128 %0, %1, off"           :: "v"(la0), "v"(ga) : "memory");
    asm volatile("global_load_async_to_lds_b128 %0, %1, off offset:16" :: "v"(la0), "v"(ga) : "memory");
    asm volatile("global_load_async_to_lds_b128 %0, %1, off"           :: "v"(lb0), "v"(gb) : "memory");
    asm volatile("global_load_async_to_lds_b128 %0, %1, off offset:16" :: "v"(lb0), "v"(gb) : "memory");
    asm volatile("global_load_async_to_lds_b128 %0, %1, off offset:32" :: "v"(lb0), "v"(gb) : "memory");
    asm volatile("global_load_async_to_lds_b128 %0, %1, off offset:48" :: "v"(lb0), "v"(gb) : "memory");
  };

  // prologue: fill buffer 0
  issue_async(0, 0);
  asm volatile("s_wait_asynccnt 0" ::: "memory");
  __syncthreads();

  for (int s = 0; s < nStages; ++s) {
    const int cur = s & 1;
    if (s + 1 < nStages) issue_async(s + 1, cur ^ 1);   // overlap next stage

    const __bf16* la = (const __bf16*)lA[cur];
    const __bf16* lb = (const __bf16*)lB[cur];
#pragma unroll
    for (int kk = 0; kk < 2; ++kk) {
      const int ko = kk * 32;
      v8bf a0lo = *(const v8bf*)&la[aOff0 + ko + hlf * 8];
      v8bf a0hi = *(const v8bf*)&la[aOff0 + ko + 16 + hlf * 8];
      v8bf a1lo = *(const v8bf*)&la[aOff1 + ko + hlf * 8];
      v8bf a1hi = *(const v8bf*)&la[aOff1 + ko + 16 + hlf * 8];
      v8bf b0lo = *(const v8bf*)&lb[bOff0 + ko + hlf * 8];
      v8bf b0hi = *(const v8bf*)&lb[bOff0 + ko + 16 + hlf * 8];
      v8bf b1lo = *(const v8bf*)&lb[bOff1 + ko + hlf * 8];
      v8bf b1hi = *(const v8bf*)&lb[bOff1 + ko + 16 + hlf * 8];
      v16bf af0, af1, bf0, bf1;
#pragma unroll
      for (int i = 0; i < 8; ++i) {
        af0[i] = a0lo[i]; af0[i + 8] = a0hi[i];
        af1[i] = a1lo[i]; af1[i + 8] = a1hi[i];
        bf0[i] = b0lo[i]; bf0[i + 8] = b0hi[i];
        bf1[i] = b1lo[i]; bf1[i + 8] = b1hi[i];
      }
      acc[0][0] = __builtin_amdgcn_wmma_f32_16x16x32_bf16(false, af0, false, bf0, (short)0, acc[0][0], false, false);
      acc[0][1] = __builtin_amdgcn_wmma_f32_16x16x32_bf16(false, af0, false, bf1, (short)0, acc[0][1], false, false);
      acc[1][0] = __builtin_amdgcn_wmma_f32_16x16x32_bf16(false, af1, false, bf0, (short)0, acc[1][0], false, false);
      acc[1][1] = __builtin_amdgcn_wmma_f32_16x16x32_bf16(false, af1, false, bf1, (short)0, acc[1][1], false, false);
    }

    asm volatile("s_wait_asynccnt 0" ::: "memory");  // next-stage loads landed
    __syncthreads();                                 // all waves done read+write
  }

  // store 32x32 wave tile (guard M tail)
#pragma unroll
  for (int mi = 0; mi < 2; ++mi) {
#pragma unroll
    for (int ni = 0; ni < 2; ++ni) {
      const int cN  = nBase + wn * 32 + ni * 16 + mrow;
      const int cM0 = mBase + wm * 32 + mi * 16 + hlf * 8;
#pragma unroll
      for (int v = 0; v < 8; ++v)
        if (cM0 + v < M) C[(size_t)(cM0 + v) * Npad + cN] = acc[mi][ni][v];
    }
  }
}

// ---------------------------------------------------------------------------
// Graph-op kernels (scatter/segment, bandwidth+atomic bound)
// ---------------------------------------------------------------------------
__device__ __forceinline__ void atomicMaxF(float* addr, float val) {
  int* ia = (int*)addr;
  int old = __float_as_int(*addr);
  while (__int_as_float(old) < val) {
    int assumed = old;
    old = atomicCAS(ia, assumed, __float_as_int(val));
    if (old == assumed) break;
  }
}

// block per edge: s[dst,:] += xl[src,:]; cnt[dst] += 1
__global__ void k_scatter_mean(const float* __restrict__ xl,
                               const int* __restrict__ src,
                               const int* __restrict__ dst,
                               float* __restrict__ s,
                               float* __restrict__ cnt, int F) {
  const int e = blockIdx.x;
  const int si = src[e], di = dst[e];
  const float* xs = xl + (size_t)si * F;
  float* sd = s + (size_t)di * F;
  for (int f = threadIdx.x; f < F; f += blockDim.x)
    atomicAdd(&sd[f], xs[f]);
  if (threadIdx.x == 0) atomicAdd(&cnt[di], 1.0f);
}

// h = relu(s/max(cnt,1) + xr + b) -> bf16
__global__ void k_sage_combine(const float* __restrict__ s,
                               const float* __restrict__ cnt,
                               const float* __restrict__ xr,
                               const float* __restrict__ bias,
                               long long M, int F,
                               unsigned short* __restrict__ out_bf) {
  long long total = M * (long long)F;
  for (long long i = blockIdx.x * (long long)blockDim.x + threadIdx.x;
       i < total; i += (long long)gridDim.x * blockDim.x) {
    long long r = i / F;
    int f = (int)(i - r * F);
    float v = s[i] / fmaxf(cnt[r], 1.0f) + xr[i] + bias[f];
    out_bf[i] = f2bf(fmaxf(v, 0.0f));
  }
}

// out = relu(C[r*ldc+f] + b[f]) -> bf16 and/or f32 (compact stride F)
__global__ void k_bias_relu(const float* __restrict__ C,
                            const float* __restrict__ bias,
                            long long M, int F, int ldc,
                            unsigned short* __restrict__ out_bf,
                            float* __restrict__ out_f32) {
  long long total = M * (long long)F;
  for (long long i = blockIdx.x * (long long)blockDim.x + threadIdx.x;
       i < total; i += (long long)gridDim.x * blockDim.x) {
    long long r = i / F;
    int f = (int)(i - r * F);
    float v = fmaxf(C[r * (long long)ldc + f] + bias[f], 0.0f);
    if (out_bf)  out_bf[i] = f2bf(v);
    if (out_f32) out_f32[i] = v;
  }
}

// concat 4 edge sets + self loops into srcAll/dstAll
__global__ void k_build_edges(const int* __restrict__ e0, const int* __restrict__ e1,
                              const int* __restrict__ e2, const int* __restrict__ e3,
                              int E, int Nn,
                              int* __restrict__ srcAll, int* __restrict__ dstAll) {
  long long total = 4LL * E + Nn;
  for (long long i = blockIdx.x * (long long)blockDim.x + threadIdx.x;
       i < total; i += (long long)gridDim.x * blockDim.x) {
    int s, d;
    if (i < E)            { s = e0[i];            d = e0[E + i]; }
    else if (i < 2LL * E) { long long j = i - E;       s = e1[j]; d = e1[E + j]; }
    else if (i < 3LL * E) { long long j = i - 2LL * E; s = e2[j]; d = e2[E + j]; }
    else if (i < 4LL * E) { long long j = i - 3LL * E; s = e3[j]; d = e3[E + j]; }
    else                  { int v = (int)(i - 4LL * E); s = v; d = v; }
    srcAll[i] = s; dstAll[i] = d;
  }
}

// alpha_s[i] = xp[i,:].a_s ; alpha_d[i] = xp[i,:].a_d
__global__ void k_alpha(const float* __restrict__ xp,
                        const float* __restrict__ a_s,
                        const float* __restrict__ a_d,
                        int M, int F,
                        float* __restrict__ as_, float* __restrict__ ad_) {
  int i = blockIdx.x * blockDim.x + threadIdx.x;
  if (i >= M) return;
  const float* row = xp + (size_t)i * F;
  float s0 = 0.0f, s1 = 0.0f;
  for (int f = 0; f < F; ++f) { float v = row[f]; s0 += v * a_s[f]; s1 += v * a_d[f]; }
  as_[i] = s0; ad_[i] = s1;
}

__global__ void k_edge_pass1(const int* __restrict__ src, const int* __restrict__ dst,
                             long long nE,
                             const float* __restrict__ as_, const float* __restrict__ ad_,
                             float* __restrict__ ebuf, float* __restrict__ emax) {
  for (long long i = blockIdx.x * (long long)blockDim.x + threadIdx.x;
       i < nE; i += (long long)gridDim.x * blockDim.x) {
    float v = as_[src[i]] + ad_[dst[i]];
    v = (v > 0.0f) ? v : NEG_SLOPE * v;
    ebuf[i] = v;
    atomicMaxF(&emax[dst[i]], v);
  }
}

__global__ void k_edge_pass2(const int* __restrict__ dst, long long nE,
                             const float* __restrict__ emax,
                             float* __restrict__ ebuf, float* __restrict__ denom) {
  for (long long i = blockIdx.x * (long long)blockDim.x + threadIdx.x;
       i < nE; i += (long long)gridDim.x * blockDim.x) {
    float ee = __expf(ebuf[i] - emax[dst[i]]);
    ebuf[i] = ee;
    atomicAdd(&denom[dst[i]], ee);
  }
}

// block per edge: out[dst,:] += (ee/denom[dst]) * xp[src,:]
__global__ void k_edge_pass3(const int* __restrict__ src, const int* __restrict__ dst,
                             const float* __restrict__ ebuf,
                             const float* __restrict__ denom,
                             const float* __restrict__ xp,
                             float* __restrict__ out, int F) {
  const long long e = blockIdx.x;
  const int si = src[e], di = dst[e];
  const float w = ebuf[e] / denom[di];
  const float* xs = xp + (size_t)si * F;
  float* od = out + (size_t)di * F;
  for (int f = threadIdx.x; f < F; f += blockDim.x)
    atomicAdd(&od[f], w * xs[f]);
}

// xl4 = h5 @ W4l.T, xr4 = h5 @ W4r.T  (64 -> 4, thread per row)
__global__ void k_small_sage_proj(const float* __restrict__ h,
                                  const float* __restrict__ Wl,
                                  const float* __restrict__ Wr,
                                  int M, int K,
                                  float* __restrict__ xl, float* __restrict__ xr) {
  int i = blockIdx.x * blockDim.x + threadIdx.x;
  if (i >= M) return;
  const float* row = h + (size_t)i * K;
  for (int o = 0; o < 4; ++o) {
    float sl = 0.0f, sr = 0.0f;
    const float* wl = Wl + o * K;
    const float* wr = Wr + o * K;
    for (int k = 0; k < K; ++k) { float v = row[k]; sl += v * wl[k]; sr += v * wr[k]; }
    xl[i * 4 + o] = sl; xr[i * 4 + o] = sr;
  }
}

__global__ void k_scatter4(const int* __restrict__ src, const int* __restrict__ dst,
                           int E, const float* __restrict__ xl,
                           float* __restrict__ s, float* __restrict__ cnt) {
  int e = blockIdx.x * blockDim.x + threadIdx.x;
  if (e >= E) return;
  int si = src[e], di = dst[e];
  for (int o = 0; o < 4; ++o) atomicAdd(&s[di * 4 + o], xl[si * 4 + o]);
  atomicAdd(&cnt[di], 1.0f);
}

// final: relu(s/max(cnt,1)+xr+b) then log_softmax over 4 -> d_out
__global__ void k_final(const float* __restrict__ s, const float* __restrict__ cnt,
                        const float* __restrict__ xr, const float* __restrict__ b,
                        int M, float* __restrict__ out) {
  int i = blockIdx.x * blockDim.x + threadIdx.x;
  if (i >= M) return;
  float c = fmaxf(cnt[i], 1.0f);
  float h[4], m = -3.0e38f;
  for (int o = 0; o < 4; ++o) {
    float v = fmaxf(s[i * 4 + o] / c + xr[i * 4 + o] + b[o], 0.0f);
    h[o] = v; m = fmaxf(m, v);
  }
  float sum = 0.0f;
  for (int o = 0; o < 4; ++o) sum += __expf(h[o] - m);
  float l = m + logf(sum);
  for (int o = 0; o < 4; ++o) out[i * 4 + o] = h[o] - l;
}

// ---------------------------------------------------------------------------
// Host-side orchestration
// ---------------------------------------------------------------------------
extern "C" void kernel_launch(void* const* d_in, const int* in_sizes, int n_in,
                              void* d_out, int out_size, void* d_ws, size_t ws_size,
                              hipStream_t stream) {
  const int Nn = 20000, Ee = 160000;
  const int FIN = 2000, FINP = 2048;      // pad K to TK multiple
  const long long nAll = 4LL * Ee + Nn;   // GAT edge count incl. self loops
  const int MB = (Nn + TM - 1) / TM;      // GEMM M-blocks (313)

  const float* x    = (const float*)d_in[0];
  const int* eIVI   = (const int*)d_in[1];
  const int* eIBI   = (const int*)d_in[2];
  const int* eIOI   = (const int*)d_in[3];
  const int* eITI   = (const int*)d_in[4];
  const float* W1l  = (const float*)d_in[5];
  const float* W1r  = (const float*)d_in[6];
  const float* b1   = (const float*)d_in[7];
  const float* Wl1  = (const float*)d_in[8];
  const float* bl1  = (const float*)d_in[9];
  const float* Wg2  = (const float*)d_in[10];
  const float* a2s  = (const float*)d_in[11];
  const float* a2d  = (const float*)d_in[12];
  const float* bg2  = (const float*)d_in[13];
  const float* Wg3  = (const float*)d_in[14];
  const float* a3s  = (const float*)d_in[15];
  const float* a3d  = (const float*)d_in[16];
  const float* bg3  = (const float*)d_in[17];
  const float* Wl2  = (const float*)d_in[18];
  const float* bl2  = (const float*)d_in[19];
  const float* W4l  = (const float*)d_in[20];
  const float* W4r  = (const float*)d_in[21];
  const float* b4   = (const float*)d_in[22];
  float* out = (float*)d_out;

  char* base = (char*)d_ws;
  size_t cur = 0;
  auto alloc = [&](size_t bytes) -> char* {
    char* p = base + cur;
    cur += (bytes + 255) & ~(size_t)255;
    return p;
  };

  unsigned short* xb    = (unsigned short*)alloc((size_t)Nn * FINP * 2);
  unsigned short* w1lb  = (unsigned short*)alloc((size_t)1024 * FINP * 2);
  unsigned short* w1rb  = (unsigned short*)alloc((size_t)1024 * FINP * 2);
  unsigned short* wl1b  = (unsigned short*)alloc((size_t)512 * 1024 * 2);
  unsigned short* wg2b  = (unsigned short*)alloc((size_t)256 * 512 * 2);
  unsigned short* wg3b  = (unsigned short*)alloc((size_t)128 * 256 * 2);
  unsigned short* wl2b  = (unsigned short*)alloc((size_t)128 * 128 * 2);   // 64 rows zero-padded to 128
  float* bufA  = (float*)alloc((size_t)Nn * 1024 * 4);
  float* bufB  = (float*)alloc((size_t)Nn * 1024 * 4);
  float* bufC  = (float*)alloc((size_t)Nn * 1024 * 4);
  unsigned short* hb = (unsigned short*)alloc((size_t)Nn * 1024 * 2); // reused bf16 staging
  int*   srcAll = (int*)alloc((size_t)nAll * 4);
  int*   dstAll = (int*)alloc((size_t)nAll * 4);
  float* ebuf   = (float*)alloc((size_t)nAll * 4);
  float* emax   = (float*)alloc((size_t)Nn * 4);
  float* denom  = (float*)alloc((size_t)Nn * 4);
  float* as_    = (float*)alloc((size_t)Nn * 4);
  float* ad_    = (float*)alloc((size_t)Nn * 4);
  float* cnt    = (float*)alloc((size_t)Nn * 4);
  float* xl4    = (float*)alloc((size_t)Nn * 4 * 4);
  float* xr4    = (float*)alloc((size_t)Nn * 4 * 4);
  float* s4     = (float*)alloc((size_t)Nn * 4 * 4);
  (void)ws_size; (void)n_in; (void)in_sizes; (void)out_size;

  const dim3 B256(256);
  const int GS = 2048;

  // ---- bf16 conversions (operands of every WMMA GEMM) ----
  k_f32_to_bf16_pad<<<GS, B256, 0, stream>>>(x,   xb,   Nn,   Nn,   FIN,  FINP);
  k_f32_to_bf16_pad<<<GS, B256, 0, stream>>>(W1l, w1lb, 1024, 1024, FIN,  FINP);
  k_f32_to_bf16_pad<<<GS, B256, 0, stream>>>(W1r, w1rb, 1024, 1024, FIN,  FINP);
  k_f32_to_bf16_pad<<<GS, B256, 0, stream>>>(Wl1, wl1b, 512,  512,  1024, 1024);
  k_f32_to_bf16_pad<<<GS, B256, 0, stream>>>(Wg2, wg2b, 256,  256,  512,  512);
  k_f32_to_bf16_pad<<<GS, B256, 0, stream>>>(Wg3, wg3b, 128,  128,  256,  256);
  k_f32_to_bf16_pad<<<GS, B256, 0, stream>>>(Wl2, wl2b, 64,   128,  128,  128);

  // ---- Stage 1: SAGEConv(2000->1024) over IVI ----
  k_gemm_bf16_wmma<<<dim3(1024 / TN, MB), B256, 0, stream>>>(xb, w1lb, bufA, Nn, 1024, FINP); // xl
  k_gemm_bf16_wmma<<<dim3(1024 / TN, MB), B256, 0, stream>>>(xb, w1rb, bufB, Nn, 1024, FINP); // xr
  k_zero_f32<<<GS, B256, 0, stream>>>(bufC, (long long)Nn * 1024);
  k_zero_f32<<<GS, B256, 0, stream>>>(cnt, Nn);
  k_scatter_mean<<<Ee, B256, 0, stream>>>(bufA, eIVI, eIVI + Ee, bufC, cnt, 1024);
  k_sage_combine<<<GS, B256, 0, stream>>>(bufC, cnt, bufB, b1, Nn, 1024, hb);  // h1 bf16

  // ---- Stage 2: Linear(1024->512) + ReLU ----
  k_gemm_bf16_wmma<<<dim3(512 / TN, MB), B256, 0, stream>>>(hb, wl1b, bufA, Nn, 512, 1024);
  k_bias_relu<<<GS, B256, 0, stream>>>(bufA, bl1, Nn, 512, 512, hb, nullptr);  // h2 bf16

  // ---- concat edges (4E + self loops) ----
  k_build_edges<<<GS, B256, 0, stream>>>(eIVI, eIBI, eIOI, eITI, Ee, Nn, srcAll, dstAll);

  // ---- Stage 3: GATConv(512->256) ----
  k_gemm_bf16_wmma<<<dim3(256 / TN, MB), B256, 0, stream>>>(hb, wg2b, bufB, Nn, 256, 512); // xp2
  k_alpha<<<(Nn + 255) / 256, B256, 0, stream>>>(bufB, a2s, a2d, Nn, 256, as_, ad_);
  k_fill_f32<<<GS, B256, 0, stream>>>(emax, -3.0e38f, Nn);
  k_zero_f32<<<GS, B256, 0, stream>>>(denom, Nn);
  k_edge_pass1<<<GS, B256, 0, stream>>>(srcAll, dstAll, nAll, as_, ad_, ebuf, emax);
  k_edge_pass2<<<GS, B256, 0, stream>>>(dstAll, nAll, emax, ebuf, denom);
  k_zero_f32<<<GS, B256, 0, stream>>>(bufC, (long long)Nn * 256);
  k_edge_pass3<<<(unsigned)nAll, dim3(128), 0, stream>>>(srcAll, dstAll, ebuf, denom, bufB, bufC, 256);
  k_bias_relu<<<GS, B256, 0, stream>>>(bufC, bg2, Nn, 256, 256, hb, nullptr);  // h3 bf16

  // ---- Stage 4: GATConv(256->128) ----
  k_gemm_bf16_wmma<<<dim3(128 / TN, MB), B256, 0, stream>>>(hb, wg3b, bufB, Nn, 128, 256); // xp3
  k_alpha<<<(Nn + 255) / 256, B256, 0, stream>>>(bufB, a3s, a3d, Nn, 128, as_, ad_);
  k_fill_f32<<<GS, B256, 0, stream>>>(emax, -3.0e38f, Nn);
  k_zero_f32<<<GS, B256, 0, stream>>>(denom, Nn);
  k_edge_pass1<<<GS, B256, 0, stream>>>(srcAll, dstAll, nAll, as_, ad_, ebuf, emax);
  k_edge_pass2<<<GS, B256, 0, stream>>>(dstAll, nAll, emax, ebuf, denom);
  k_zero_f32<<<GS, B256, 0, stream>>>(bufC, (long long)Nn * 128);
  k_edge_pass3<<<(unsigned)nAll, dim3(128), 0, stream>>>(srcAll, dstAll, ebuf, denom, bufB, bufC, 128);
  k_bias_relu<<<GS, B256, 0, stream>>>(bufC, bg3, Nn, 128, 128, hb, nullptr);  // h4 bf16

  // ---- Stage 5: Linear(128->64) + ReLU (N padded 64->128, ldc=128) ----
  k_gemm_bf16_wmma<<<dim3(128 / TN, MB), B256, 0, stream>>>(hb, wl2b, bufA, Nn, 128, 128);
  k_bias_relu<<<GS, B256, 0, stream>>>(bufA, bl2, Nn, 64, 128, nullptr, bufB); // h5 f32

  // ---- Stage 6: SAGEConv(64->4) over IOI + log_softmax ----
  k_small_sage_proj<<<(Nn + 255) / 256, B256, 0, stream>>>(bufB, W4l, W4r, Nn, 64, xl4, xr4);
  k_zero_f32<<<GS, B256, 0, stream>>>(s4, (long long)Nn * 4);
  k_zero_f32<<<GS, B256, 0, stream>>>(cnt, Nn);
  k_scatter4<<<(Ee + 255) / 256, B256, 0, stream>>>(eIOI, eIOI + Ee, Ee, xl4, s4, cnt);
  k_final<<<(Nn + 255) / 256, B256, 0, stream>>>(s4, cnt, xr4, b4, Nn, out);
}
